// RRB_body_27032524161679
// MI455X (gfx1250) — matrix operand order repaired and checked by reading, not compiled, not run on hardware
//
#include <hip/hip_runtime.h>

typedef _Float16 half_t;
typedef __attribute__((ext_vector_type(16))) _Float16 v16h;
typedef __attribute__((ext_vector_type(8)))  _Float16 v8h;
typedef __attribute__((ext_vector_type(8)))  float    v8f;

#define WMMA_F16(a,b,c) __builtin_amdgcn_wmma_f32_16x16x32_f16(false,(a),false,(b),(short)0,(c),false,false)

static __device__ __forceinline__ v16h make_v16(v8h lo, v8h hi){
  v16h r;
#pragma unroll
  for (int i = 0; i < 8; i++){ r[i] = lo[i]; r[8+i] = hi[i]; }
  return r;
}

// ---------------------------------------------------------------------------
// CDNA5 async global->LDS copy (tracked by ASYNCcnt, no VGPR round trip).
// LDS operand = low 32 bits of the generic shared pointer (aperture mapping).
// ---------------------------------------------------------------------------
static __device__ __forceinline__ void async_copy_b128(void* lds_dst, const void* gsrc){
  unsigned l = (unsigned)(unsigned long long)(uintptr_t)lds_dst;
  asm volatile("global_load_async_to_lds_b128 %0, %1, off"
               :: "v"(l), "v"(gsrc) : "memory");
}
static __device__ __forceinline__ void async_wait0(){
  asm volatile("s_wait_asynccnt 0x0" ::: "memory");
}

// ---------------------------------------------------------------------------
// Weight packing: f32 HWIO -> f16 per-lane WMMA B-fragments.
// B layout (32x16, f16): lane<16 holds N=lane, K = (h&7) + (h>=8 ? 16 : 0) (+8 for lane>=16).
// dst index = ((ks*NT + nt)*32 + lane)*16 + h
// ---------------------------------------------------------------------------
__global__ void pack_weights_kernel(const float* __restrict__ src, half_t* __restrict__ dst,
                                    int taps, int cin, int cout, int KS, int NT){
  int total = KS * NT * 512;
  int idx = blockIdx.x * 256 + threadIdx.x;
  if (idx >= total) return;
  int h    = idx & 15;
  int lane = (idx >> 4) & 31;
  int nt   = (idx >> 9) % NT;
  int ks   = idx / (512 * NT);
  int n = nt * 16 + (lane & 15);
  int k = ks * 32 + (h & 7) + ((h & 8) ? 16 : 0) + ((lane & 16) ? 8 : 0);
  float v = 0.f;
  if (k < taps * cin){
    int tap = k / cin, c = k % cin;
    v = src[(size_t)(tap * cin + c) * cout + n];
  }
  dst[idx] = (half_t)v;
}

// ---------------------------------------------------------------------------
// Generic 3x3 stride-1 SAME conv via WMMA. One block = 64-pixel row strip.
// Halo tile staged into LDS with async-to-LDS copies.
// MASK: fea = relu(conv) * (chm1[c]*spa[p] + chm0[c])   (body layers)
// ---------------------------------------------------------------------------
template<int CIN, int COUT, int NW, bool MASK>
__global__ __launch_bounds__(NW*32)
void conv3x3_kernel(const half_t* __restrict__ in, const half_t* __restrict__ wp,
                    const float* __restrict__ bias, const half_t* __restrict__ spa,
                    const float* __restrict__ chm, half_t* __restrict__ out,
                    int H, int W){
  constexpr int TW = 64;
  constexpr int NT = COUT / 16;
  constexpr int KS = (9 * CIN + 31) / 32;
  constexpr int CG = CIN / 8;
  __shared__ half_t in_s[3 * (TW + 2) * CIN];
  __shared__ half_t spa_s[TW];

  int s   = blockIdx.x;
  int wpr = W / TW;
  int w0  = (s % wpr) * TW;
  int h   = (s / wpr) % H;
  int b   = s / (wpr * H);
  int tid = threadIdx.x;
  const int nth = NW * 32;

  for (int u = tid; u < 3 * (TW + 2) * CG; u += nth){
    int cg  = u % CG;
    int col = (u / CG) % (TW + 2);
    int ry  = u / (CG * (TW + 2));
    int hin = h + ry - 1, win = w0 + col - 1;
    half_t* dst = in_s + ((size_t)ry * (TW + 2) + col) * CIN + cg * 8;
    if (hin >= 0 && hin < H && win >= 0 && win < W){
      async_copy_b128(dst, in + ((size_t)(b * H + hin) * W + win) * CIN + cg * 8);
    } else {
      v8h z = {};
      *(v8h*)dst = z;
    }
  }
  if (MASK){
    for (int u = tid; u < TW; u += nth)
      spa_s[u] = spa[(size_t)(b * H + h) * W + w0 + u];
  }
  async_wait0();
  __syncthreads();

  int lane = tid & 31;
  int wave = tid >> 5;
  int mlo  = lane & 15;
  int hk   = (lane & 16) ? 8 : 0;

  for (int t = wave; t < 4 * NT; t += NW){
    int msub = (t & 3) * 16;
    int nt   = t >> 2;
    int n    = nt * 16 + mlo;
    v8f acc = {};
    const half_t* wptr = wp + ((size_t)nt * 32 + lane) * 16;
    for (int ks = 0; ks < KS; ks++){
      v8h r0 = {}, r1 = {};
      int k0 = ks * 32 + hk;
      int k1 = k0 + 16;
      { int tap = k0 / CIN;
        if (tap < 9){
          int ky = tap / 3, kx = tap % 3, c = k0 % CIN;
          r0 = *(const v8h*)(in_s + ((size_t)ky * (TW + 2) + (msub + mlo) + kx) * CIN + c);
        } }
      { int tap = k1 / CIN;
        if (tap < 9){
          int ky = tap / 3, kx = tap % 3, c = k1 % CIN;
          r1 = *(const v8h*)(in_s + ((size_t)ky * (TW + 2) + (msub + mlo) + kx) * CIN + c);
        } }
      v16h a = make_v16(r0, r1);
      const half_t* wk = wptr + (size_t)ks * NT * 512;
      v16h bf = make_v16(*(const v8h*)wk, *(const v8h*)(wk + 8));
      acc = WMMA_F16(a, bf, acc);
    }
    float bn = bias[n];
    float c0 = 0.f, c1 = 0.f;
    if (MASK){ c0 = chm[n * 2 + 0]; c1 = chm[n * 2 + 1]; }
    int mbase = msub + ((lane & 16) ? 8 : 0);
#pragma unroll
    for (int v = 0; v < 8; v++){
      int m = mbase + v;
      float f = acc[v] + bn;
      f = fmaxf(f, 0.f);
      if (MASK) f *= (c1 * (float)spa_s[m] + c0);
      out[((size_t)(b * H + h) * W + w0 + m) * COUT + n] = (half_t)f;
    }
  }
}

// ---------------------------------------------------------------------------
// down: 3x3 stride-2 SAME conv, f32 input [4,512,512,32] -> relu f16 [4,256,256,64]
// jax SAME for s=2,k=3,in=512: pad_lo=0 -> input index = 2*out + tap
// (VGPR staging kept: converts f32 -> f16 en route)
// ---------------------------------------------------------------------------
template<int NW>
__global__ __launch_bounds__(NW*32)
void downconv_kernel(const float* __restrict__ x, const half_t* __restrict__ wp,
                     const float* __restrict__ bias, half_t* __restrict__ out){
  constexpr int CIN = 32, COUT = 64, NT = 4, KS = 9, TW = 64;
  constexpr int H = 256, W = 256, HIN = 512, WIN = 512;
  constexpr int LW = 130;          // 129 needed + pad
  __shared__ half_t in_s[3 * LW * CIN];

  int s   = blockIdx.x;
  int wpr = W / TW;                // 4
  int w0  = (s % wpr) * TW;
  int h   = (s / wpr) % H;
  int b   = s / (wpr * H);
  int tid = threadIdx.x;
  const int nth = NW * 32;

  for (int u = tid; u < 3 * LW * (CIN / 8); u += nth){
    int cg  = u % (CIN / 8);
    int col = (u / (CIN / 8)) % LW;
    int ry  = u / ((CIN / 8) * LW);
    int hin = 2 * h + ry, win = 2 * w0 + col;
    v8h v = {};
    if (col < 129 && hin < HIN && win < WIN){
      const float* p = x + ((size_t)(b * HIN + hin) * WIN + win) * CIN + cg * 8;
#pragma unroll
      for (int i = 0; i < 8; i++) v[i] = (half_t)p[i];
    }
    *(v8h*)(in_s + ((size_t)ry * LW + col) * CIN + cg * 8) = v;
  }
  __syncthreads();

  int lane = tid & 31;
  int wave = tid >> 5;
  int mlo  = lane & 15;
  int hk   = (lane & 16) ? 8 : 0;

  for (int t = wave; t < 4 * NT; t += NW){
    int msub = (t & 3) * 16;
    int nt   = t >> 2;
    int n    = nt * 16 + mlo;
    v8f acc = {};
    const half_t* wptr = wp + ((size_t)nt * 32 + lane) * 16;
    for (int ks = 0; ks < KS; ks++){
      int k0 = ks * 32 + hk;
      int k1 = k0 + 16;
      int ty0 = k0 / CIN, ty1 = k1 / CIN;
      v8h r0 = *(const v8h*)(in_s + ((size_t)(ty0 / 3) * LW + 2 * (msub + mlo) + (ty0 % 3)) * CIN + (k0 % CIN));
      v8h r1 = *(const v8h*)(in_s + ((size_t)(ty1 / 3) * LW + 2 * (msub + mlo) + (ty1 % 3)) * CIN + (k1 % CIN));
      v16h a = make_v16(r0, r1);
      const half_t* wk = wptr + (size_t)ks * NT * 512;
      v16h bf = make_v16(*(const v8h*)wk, *(const v8h*)(wk + 8));
      acc = WMMA_F16(a, bf, acc);
    }
    float bn = bias[n];
    int mbase = msub + ((lane & 16) ? 8 : 0);
#pragma unroll
    for (int v = 0; v < 8; v++){
      int m = mbase + v;
      float f = fmaxf(acc[v] + bn, 0.f);
      out[((size_t)(b * H + h) * W + w0 + m) * COUT + n] = (half_t)f;
    }
  }
}

// ---------------------------------------------------------------------------
// 1x1 GEMM helper (A staged in LDS row-major [64][K])
// ---------------------------------------------------------------------------
template<int K, int NT>
__device__ __forceinline__ void gemm1x1_tile(const half_t* a_s, const half_t* wp,
                                             int msub, int nt, int lane, v8f& acc){
  int mlo = lane & 15;
  int hk  = (lane & 16) ? 8 : 0;
  const half_t* wptr = wp + ((size_t)nt * 32 + lane) * 16;
  constexpr int KS = K / 32;
  for (int ks = 0; ks < KS; ks++){
    int k0 = ks * 32 + hk;
    const half_t* ap = a_s + (size_t)(msub + mlo) * K;
    v16h a  = make_v16(*(const v8h*)(ap + k0), *(const v8h*)(ap + k0 + 16));
    const half_t* wk = wptr + (size_t)ks * NT * 512;
    v16h bf = make_v16(*(const v8h*)wk, *(const v8h*)(wk + 8));
    acc = WMMA_F16(a, bf, acc);
  }
}

// collect: 1x1 conv over concat(b0,b1,b2) K=192 -> 64, +bias (no relu)
// A staging is a pure copy -> async-to-LDS
template<int NW>
__global__ __launch_bounds__(NW*32)
void collect_kernel(const half_t* __restrict__ b0, const half_t* __restrict__ b1,
                    const half_t* __restrict__ b2, const half_t* __restrict__ wp,
                    const float* __restrict__ bias, half_t* __restrict__ out){
  constexpr int K = 192, NT = 4, TM = 64;
  __shared__ half_t a_s[TM * K];
  int p0  = blockIdx.x * TM;
  int tid = threadIdx.x;
  const int nth = NW * 32;
  for (int u = tid; u < TM * (K / 8); u += nth){
    int cg = u % (K / 8);
    int px = u / (K / 8);
    const half_t* src = (cg < 8) ? b0 : (cg < 16 ? b1 : b2);
    int c = (cg & 7) * 8;
    async_copy_b128(a_s + (size_t)px * K + cg * 8, src + (size_t)(p0 + px) * 64 + c);
  }
  async_wait0();
  __syncthreads();
  int lane = tid & 31, wave = tid >> 5;
  for (int t = wave; t < 4 * NT; t += NW){
    int msub = (t & 3) * 16, nt = t >> 2;
    int n = nt * 16 + (lane & 15);
    v8f acc = {};
    gemm1x1_tile<K, NT>(a_s, wp, msub, nt, lane, acc);
    float bn = bias[n];
    int mbase = msub + ((lane & 16) ? 8 : 0);
#pragma unroll
    for (int v = 0; v < 8; v++)
      out[(size_t)(p0 + mbase + v) * 64 + n] = (half_t)(acc[v] + bn);
  }
}

// c2c: fused out1 = out0 + tr[b] + tensor; A = [out1 | y], K=128 -> 64 f32 output
template<int NW>
__global__ __launch_bounds__(NW*32)
void c2c_kernel(const half_t* __restrict__ out0, const half_t* __restrict__ tens,
                const float* __restrict__ tr, const float* __restrict__ y,
                const half_t* __restrict__ wp, const float* __restrict__ bias,
                float* __restrict__ out){
  constexpr int K = 128, NT = 4, TM = 64;
  __shared__ half_t a_s[TM * K];
  int p0  = blockIdx.x * TM;
  int b   = p0 >> 16;              // 65536 pixels / batch
  int tid = threadIdx.x;
  const int nth = NW * 32;
  for (int u = tid; u < TM * (K / 8); u += nth){
    int cg = u & 15;
    int px = u >> 4;
    size_t gp = (size_t)p0 + px;
    v8h v;
    if (cg < 8){
      int c = cg * 8;
      v8h a = *(const v8h*)(out0 + gp * 64 + c);
      v8h t = *(const v8h*)(tens + gp * 64 + c);
#pragma unroll
      for (int i = 0; i < 8; i++)
        v[i] = (half_t)((float)a[i] + (float)t[i] + tr[b * 64 + c + i]);
    } else {
      const float* yp = y + gp * 64 + (cg - 8) * 8;
#pragma unroll
      for (int i = 0; i < 8; i++) v[i] = (half_t)yp[i];
    }
    *(v8h*)(a_s + (size_t)px * K + cg * 8) = v;
  }
  __syncthreads();
  int lane = tid & 31, wave = tid >> 5;
  for (int t = wave; t < 4 * NT; t += NW){
    int msub = (t & 3) * 16, nt = t >> 2;
    int n = nt * 16 + (lane & 15);
    v8f acc = {};
    gemm1x1_tile<K, NT>(a_s, wp, msub, nt, lane, acc);
    float bn = bias[n];
    int mbase = msub + ((lane & 16) ? 8 : 0);
#pragma unroll
    for (int v = 0; v < 8; v++)
      out[(size_t)(p0 + mbase + v) * 64 + n] = acc[v] + bn;
  }
}

// ---------------------------------------------------------------------------
// Small fused kernels (mask branch / gumbel / GC block)
// ---------------------------------------------------------------------------
__global__ void avgpool_kernel(const half_t* __restrict__ t0, half_t* __restrict__ pooled){
  int idx = blockIdx.x * 256 + threadIdx.x;          // 4*128*128*16
  if (idx >= 4 * 128 * 128 * 16) return;
  int c  = idx & 15;
  int pw = (idx >> 4) & 127;
  int ph = (idx >> 11) & 127;
  int b  = idx >> 18;
  size_t base = ((size_t)(b * 256 + 2 * ph) * 256 + 2 * pw) * 16 + c;
  float s = (float)t0[base] + (float)t0[base + 16]
          + (float)t0[base + 256 * 16] + (float)t0[base + 256 * 16 + 16];
  pooled[idx] = (half_t)(0.25f * s);
}

// conv_transpose(3x3, stride2, VALID) + bias + relu + gumbel-softmax + slice -> spa
__global__ void smT_spa_kernel(const half_t* __restrict__ t3, const float* __restrict__ w,
                               const float* __restrict__ bias, const float* __restrict__ g,
                               half_t* __restrict__ spa){
  int idx = blockIdx.x * 256 + threadIdx.x;          // 4*256*256
  int ox = (idx & 255) + 1;
  int oy = ((idx >> 8) & 255) + 1;
  int b  = idx >> 16;
  float v0 = bias[0], v1 = bias[1];
  for (int ky = 0; ky < 3; ky++){
    int iy = oy - ky;
    if (iy < 0 || (iy & 1)) continue;
    iy >>= 1; if (iy >= 128) continue;
    for (int kx = 0; kx < 3; kx++){
      int ix = ox - kx;
      if (ix < 0 || (ix & 1)) continue;
      ix >>= 1; if (ix >= 128) continue;
      const half_t* tp = t3 + ((size_t)(b * 128 + iy) * 128 + ix) * 16;
      const float*  wp = w + (size_t)(ky * 3 + kx) * 32;
#pragma unroll
      for (int ci = 0; ci < 16; ci++){
        float a = (float)tp[ci];
        v0 += a * wp[ci * 2 + 0];
        v1 += a * wp[ci * 2 + 1];
      }
    }
  }
  v0 = fmaxf(v0, 0.f); v1 = fmaxf(v1, 0.f);
  const float* gp = g + ((size_t)(b * 257 + oy) * 257 + ox) * 2;
  v0 += gp[0]; v1 += gp[1];                          // tau = 1
  float m = fmaxf(v0, v1);
  float e0 = __expf(v0 - m), e1 = __expf(v1 - m);
  spa[(size_t)(b * 256 + (oy - 1)) * 256 + (ox - 1)] = (half_t)(e1 / (e0 + e1));
}

__global__ void chmask_kernel(const float* __restrict__ logits, const float* __restrict__ g,
                              float* __restrict__ chm){
  int idx = blockIdx.x * 64 + threadIdx.x;           // 3*64
  if (idx >= 192) return;
  float l0 = logits[idx * 2]     + g[idx * 2];
  float l1 = logits[idx * 2 + 1] + g[idx * 2 + 1];
  float m = fmaxf(l0, l1);
  float e0 = __expf(l0 - m), e1 = __expf(l1 - m), inv = 1.f / (e0 + e1);
  chm[idx * 2] = e0 * inv; chm[idx * 2 + 1] = e1 * inv;
}

__global__ void gc_logits_kernel(const half_t* __restrict__ out0, const float* __restrict__ key,
                                 float* __restrict__ l){
  int p = blockIdx.x * 256 + threadIdx.x;            // NPIX
  const half_t* s = out0 + (size_t)p * 64;
  float acc = 0.f;
#pragma unroll
  for (int c = 0; c < 64; c++) acc += (float)s[c] * key[c];
  l[p] = acc;
}

__global__ void gc_stats_kernel(const float* __restrict__ l, float* __restrict__ stats){
  __shared__ float red[256];
  int b = blockIdx.x, tid = threadIdx.x;
  const float* lb = l + (size_t)b * 65536;
  float mx = -1e30f;
  for (int p = tid; p < 65536; p += 256) mx = fmaxf(mx, lb[p]);
  red[tid] = mx; __syncthreads();
  for (int s = 128; s > 0; s >>= 1){ if (tid < s) red[tid] = fmaxf(red[tid], red[tid + s]); __syncthreads(); }
  mx = red[0]; __syncthreads();
  float sum = 0.f;
  for (int p = tid; p < 65536; p += 256) sum += __expf(lb[p] - mx);
  red[tid] = sum; __syncthreads();
  for (int s = 128; s > 0; s >>= 1){ if (tid < s) red[tid] += red[tid + s]; __syncthreads(); }
  if (tid == 0){ stats[b * 2] = mx; stats[b * 2 + 1] = red[0]; }
}

// deterministic two-stage attention pooling: partials per 1024-pixel block
__global__ void gc_ctx_part_kernel(const float* __restrict__ l, const float* __restrict__ stats,
                                   const half_t* __restrict__ out0, float* __restrict__ part){
  __shared__ float wsh[1024];
  __shared__ float red2[256];
  int blk = blockIdx.x;                              // 256 blocks
  int p0  = blk * 1024;
  int b   = p0 >> 16;
  int tid = threadIdx.x;
  float mx = stats[b * 2], inv = 1.f / stats[b * 2 + 1];
  for (int i = tid; i < 1024; i += 256) wsh[i] = __expf(l[p0 + i] - mx) * inv;
  __syncthreads();
  int c = tid & 63, sub = tid >> 6;                  // 4 pixel sub-lanes per channel
  float acc = 0.f;
  for (int i = sub; i < 1024; i += 4)
    acc += wsh[i] * (float)out0[(size_t)(p0 + i) * 64 + c];
  red2[tid] = acc; __syncthreads();
  if (sub == 0)
    part[(size_t)blk * 64 + c] = red2[c] + red2[64 + c] + red2[128 + c] + red2[192 + c];
}

__global__ void gc_ctx_reduce_kernel(const float* __restrict__ part, float* __restrict__ ctx){
  int b = blockIdx.x, c = threadIdx.x;               // 4 x 64
  float acc = 0.f;
  for (int j = 0; j < 64; j++) acc += part[(size_t)(b * 64 + j) * 64 + c];
  ctx[b * 64 + c] = acc;
}

__global__ void gc_mlp_kernel(const float* __restrict__ ctx, const float* __restrict__ w1,
                              const float* __restrict__ bb1, const float* __restrict__ ln_g,
                              const float* __restrict__ ln_b, const float* __restrict__ w2,
                              const float* __restrict__ bb2, float* __restrict__ tr){
  __shared__ float h8[8];
  __shared__ float st[2];
  int b = blockIdx.x, tid = threadIdx.x;             // 4 x 64
  if (tid < 8){
    float a = bb1[tid];
    for (int c = 0; c < 64; c++) a += ctx[b * 64 + c] * w1[c * 8 + tid];
    h8[tid] = a;
  }
  __syncthreads();
  if (tid == 0){
    float mu = 0.f;  for (int j = 0; j < 8; j++) mu += h8[j];  mu *= 0.125f;
    float var = 0.f; for (int j = 0; j < 8; j++){ float d = h8[j] - mu; var += d * d; } var *= 0.125f;
    st[0] = mu; st[1] = rsqrtf(var + 1e-5f);
  }
  __syncthreads();
  float acc = bb2[tid];
  for (int j = 0; j < 8; j++){
    float hj = (h8[j] - st[0]) * st[1] * ln_g[j] + ln_b[j];
    hj = fmaxf(hj, 0.f);
    acc += hj * w2[j * 64 + tid];
  }
  tr[b * 64 + tid] = acc;
}

// ---------------------------------------------------------------------------
// Host orchestration
// ---------------------------------------------------------------------------
extern "C" void kernel_launch(void* const* d_in, const int* in_sizes, int n_in,
                              void* d_out, int out_size, void* d_ws, size_t ws_size,
                              hipStream_t stream){
  (void)in_sizes; (void)n_in; (void)out_size; (void)ws_size;
  const float* x      = (const float*)d_in[0];
  const float* y      = (const float*)d_in[1];
  const float* down_w = (const float*)d_in[2];  const float* down_b = (const float*)d_in[3];
  const float* sm0_w  = (const float*)d_in[4];  const float* sm0_b  = (const float*)d_in[5];
  const float* sm2_w  = (const float*)d_in[6];  const float* sm2_b  = (const float*)d_in[7];
  const float* sm3_w  = (const float*)d_in[8];  const float* sm3_b  = (const float*)d_in[9];
  const float* smT_w  = (const float*)d_in[10]; const float* smT_b  = (const float*)d_in[11];
  const float* bw0 = (const float*)d_in[12]; const float* bb0 = (const float*)d_in[13];
  const float* bw1 = (const float*)d_in[14]; const float* bb1 = (const float*)d_in[15];
  const float* bw2 = (const float*)d_in[16]; const float* bb2 = (const float*)d_in[17];
  const float* ch_logits = (const float*)d_in[18];
  const float* col_w = (const float*)d_in[19]; const float* col_b = (const float*)d_in[20];
  const float* gc_key = (const float*)d_in[21];
  const float* gc_t1w = (const float*)d_in[22]; const float* gc_t1b = (const float*)d_in[23];
  const float* gc_lng = (const float*)d_in[24]; const float* gc_lnb = (const float*)d_in[25];
  const float* gc_t2w = (const float*)d_in[26]; const float* gc_t2b = (const float*)d_in[27];
  const float* c2c_w = (const float*)d_in[28]; const float* c2c_b = (const float*)d_in[29];
  const float* g_spa = (const float*)d_in[30]; const float* g_ch  = (const float*)d_in[31];
  float* out = (float*)d_out;

  const size_t NPIX = 4ull * 256 * 256;
  char* base = (char*)d_ws;
  size_t off = 0;
  auto alloc = [&](size_t bytes) -> void* {
    void* p = base + off;
    off = (off + bytes + 255) & ~(size_t)255;
    return p;
  };
  half_t* tensor_f16 = (half_t*)alloc(NPIX * 64 * 2);
  half_t* fb0        = (half_t*)alloc(NPIX * 64 * 2);
  half_t* fb1        = (half_t*)alloc(NPIX * 64 * 2);
  half_t* fb2        = (half_t*)alloc(NPIX * 64 * 2);
  half_t* out0_f16   = (half_t*)alloc(NPIX * 64 * 2);
  half_t* t0_f16     = (half_t*)alloc(NPIX * 16 * 2);
  half_t* pooled     = (half_t*)alloc(65536ull * 16 * 2);
  half_t* t2_f16     = (half_t*)alloc(65536ull * 16 * 2);
  half_t* t3_f16     = (half_t*)alloc(65536ull * 16 * 2);
  half_t* spa_f16    = (half_t*)alloc(NPIX * 2);
  float*  chm        = (float*)alloc(192 * 2 * 4);
  float*  gcl        = (float*)alloc(NPIX * 4);
  float*  gcstats    = (float*)alloc(8 * 4);
  float*  gcpart     = (float*)alloc(256ull * 64 * 4);
  float*  gcctx      = (float*)alloc(256 * 4);
  float*  gctr       = (float*)alloc(256 * 4);
  half_t* wp_down = (half_t*)alloc(9  * 4 * 512 * 2);
  half_t* wp_sm0  = (half_t*)alloc(18 * 1 * 512 * 2);
  half_t* wp_sm2  = (half_t*)alloc(5  * 1 * 512 * 2);
  half_t* wp_sm3  = (half_t*)alloc(5  * 1 * 512 * 2);
  half_t* wp_b0   = (half_t*)alloc(18 * 4 * 512 * 2);
  half_t* wp_b1   = (half_t*)alloc(18 * 4 * 512 * 2);
  half_t* wp_b2   = (half_t*)alloc(18 * 4 * 512 * 2);
  half_t* wp_col  = (half_t*)alloc(6  * 4 * 512 * 2);
  half_t* wp_c2c  = (half_t*)alloc(4  * 4 * 512 * 2);

  auto pack = [&](const float* src, half_t* dst, int taps, int cin, int cout){
    int KS = (taps * cin + 31) / 32, NT = cout / 16;
    int total = KS * NT * 512;
    pack_weights_kernel<<<(total + 255) / 256, 256, 0, stream>>>(src, dst, taps, cin, cout, KS, NT);
  };
  pack(down_w, wp_down, 9, 32, 64);
  pack(sm0_w,  wp_sm0,  9, 64, 16);
  pack(sm2_w,  wp_sm2,  9, 16, 16);
  pack(sm3_w,  wp_sm3,  9, 16, 16);
  pack(bw0,    wp_b0,   9, 64, 64);
  pack(bw1,    wp_b1,   9, 64, 64);
  pack(bw2,    wp_b2,   9, 64, 64);
  pack(col_w,  wp_col,  1, 192, 64);
  pack(c2c_w,  wp_c2c,  1, 128, 64);

  chmask_kernel<<<3, 64, 0, stream>>>(ch_logits, g_ch, chm);

  // down conv (WMMA): x -> tensor_f16
  downconv_kernel<8><<<4096, 256, 0, stream>>>(x, wp_down, down_b, tensor_f16);

  // spatial mask branch
  conv3x3_kernel<64, 16, 4, false><<<4096, 128, 0, stream>>>(
      tensor_f16, wp_sm0, sm0_b, nullptr, nullptr, t0_f16, 256, 256);
  avgpool_kernel<<<4096, 256, 0, stream>>>(t0_f16, pooled);
  conv3x3_kernel<16, 16, 4, false><<<1024, 128, 0, stream>>>(
      pooled, wp_sm2, sm2_b, nullptr, nullptr, t2_f16, 128, 128);
  conv3x3_kernel<16, 16, 4, false><<<1024, 128, 0, stream>>>(
      t2_f16, wp_sm3, sm3_b, nullptr, nullptr, t3_f16, 128, 128);
  smT_spa_kernel<<<1024, 256, 0, stream>>>(t3_f16, smT_w, smT_b, g_spa, spa_f16);

  // SMB body (WMMA, fused relu + channel/spatial gumbel masks)
  conv3x3_kernel<64, 64, 8, true><<<4096, 256, 0, stream>>>(
      tensor_f16, wp_b0, bb0, spa_f16, chm + 0 * 128, fb0, 256, 256);
  conv3x3_kernel<64, 64, 8, true><<<4096, 256, 0, stream>>>(
      fb0, wp_b1, bb1, spa_f16, chm + 1 * 128, fb1, 256, 256);
  conv3x3_kernel<64, 64, 8, true><<<4096, 256, 0, stream>>>(
      fb1, wp_b2, bb2, spa_f16, chm + 2 * 128, fb2, 256, 256);

  // collect 1x1 (WMMA)
  collect_kernel<8><<<4096, 256, 0, stream>>>(fb0, fb1, fb2, wp_col, col_b, out0_f16);

  // global context block
  gc_logits_kernel<<<1024, 256, 0, stream>>>(out0_f16, gc_key, gcl);
  gc_stats_kernel<<<4, 256, 0, stream>>>(gcl, gcstats);
  gc_ctx_part_kernel<<<256, 256, 0, stream>>>(gcl, gcstats, out0_f16, gcpart);
  gc_ctx_reduce_kernel<<<4, 64, 0, stream>>>(gcpart, gcctx);
  gc_mlp_kernel<<<4, 64, 0, stream>>>(gcctx, gc_t1w, gc_t1b, gc_lng, gc_lnb, gc_t2w, gc_t2b, gctr);

  // fused residual + concat(y) + 1x1 conv (WMMA) -> f32 output
  c2c_kernel<8><<<4096, 256, 0, stream>>>(out0_f16, tensor_f16, gctr, y, wp_c2c, c2c_b, out);
}
